// CausalInferenceEngine_21096879358444
// MI455X (gfx1250) — matrix operand order repaired and checked by reading, not compiled, not run on hardware
//
#include <hip/hip_runtime.h>

// ---------------------------------------------------------------------------
// CDNA5 (gfx1250) fused CausalInferenceEngine
//   B=32, V=48, I=24, H=768
//   All GEMMs via v_wmma_f32_16x16x32_bf16 (bf16 operands, f32 accum).
// ---------------------------------------------------------------------------

#define Bc 32
#define Vc 48
#define Ic 24
#define Hc 768

typedef __attribute__((ext_vector_type(4)))  unsigned int u32x4;
typedef __attribute__((ext_vector_type(16))) __bf16       bf16x16;
typedef __attribute__((ext_vector_type(8)))  float        f32x8;

union FragAB {
  u32x4   q[2];   // 8 dwords = 16 bf16
  bf16x16 v;
};

__device__ __forceinline__ unsigned short f32_to_bf16(float f) {
  unsigned int u = __float_as_uint(f);
  unsigned int r = u + 0x7FFFu + ((u >> 16) & 1u);   // round-to-nearest-even
  return (unsigned short)(r >> 16);
}

// ---------------- elementwise f32 -> bf16 ----------------------------------
__global__ void k_cvt_bf16(const float* __restrict__ src,
                           unsigned short* __restrict__ dst, int n) {
  int idx = blockIdx.x * 256 + threadIdx.x;
  if (idx < n) dst[idx] = f32_to_bf16(src[idx]);
}

// ---------------- weight [K,N] f32 -> [N,K] bf16 (transpose) ---------------
__global__ void k_wt_bf16(const float* __restrict__ src,
                          unsigned short* __restrict__ dst, int K, int N) {
  int idx = blockIdx.x * 256 + threadIdx.x;
  if (idx >= K * N) return;
  int n = idx / K, k = idx % K;
  dst[idx] = f32_to_bf16(src[(size_t)k * N + n]);
}

// ---------------- generic WMMA GEMM: C[M,N] = A[M,K] @ Bt[N,K]^T -----------
// 1 wave per block; each wave computes a 32(M) x 64(N) tile.
__global__ void k_gemm(const unsigned short* __restrict__ A,   // [M,K] bf16
                       const unsigned short* __restrict__ Bt,  // [N,K] bf16
                       float* __restrict__ C, int M, int N, int K) {
  const int lane  = threadIdx.x & 31;
  const int mrow  = lane & 15;
  const int half  = lane >> 4;
  const int tileN = blockIdx.x * 64;
  const int tileM = blockIdx.y * 32;

  f32x8 acc[2][4] = {};

  for (int kk = 0; kk < K; kk += 32) {
    FragAB af[2];
#pragma unroll
    for (int mi = 0; mi < 2; mi++) {
      // A 16x32 bf16 layout: lane m holds K = 2r + 8*half (r=0..3), +16 (r=4..7)
      const unsigned short* ap =
          A + (size_t)(tileM + mi * 16 + mrow) * K + kk + 8 * half;
      af[mi].q[0] = *(const u32x4*)(ap);
      af[mi].q[1] = *(const u32x4*)(ap + 16);
    }
    FragAB bfr[4];
#pragma unroll
    for (int ni = 0; ni < 4; ni++) {
      // B 32x16 bf16 layout: lane n holds K = 2r + 8*half (r=0..3), +16 (r=4..7)
      const unsigned short* bp =
          Bt + (size_t)(tileN + ni * 16 + mrow) * K + kk + 8 * half;
      bfr[ni].q[0] = *(const u32x4*)(bp);
      bfr[ni].q[1] = *(const u32x4*)(bp + 16);
    }
#pragma unroll
    for (int mi = 0; mi < 2; mi++)
#pragma unroll
      for (int ni = 0; ni < 4; ni++)
        acc[mi][ni] = __builtin_amdgcn_wmma_f32_16x16x32_bf16(
            false, af[mi].v, false, bfr[ni].v, (short)0, acc[mi][ni],
            false, false);
  }

#pragma unroll
  for (int mi = 0; mi < 2; mi++)
#pragma unroll
    for (int ni = 0; ni < 4; ni++)
#pragma unroll
      for (int j = 0; j < 8; j++) {
        int row = tileM + mi * 16 + j + 8 * half;  // C/D: VGPR j -> M=j(+8)
        int col = tileN + ni * 16 + mrow;
        C[(size_t)row * N + col] = acc[mi][ni][j];
      }
}

// ---------------- LN(+bias)+ReLU for enc; wave per row ---------------------
__global__ void k_ln_relu_enc(const float* __restrict__ X,
                              const float* __restrict__ bias,
                              const float* __restrict__ g,
                              const float* __restrict__ be,
                              float* __restrict__ outF,
                              unsigned short* __restrict__ outB) {
  const int wave = threadIdx.x >> 5, lane = threadIdx.x & 31;
  const int row  = blockIdx.x * 8 + wave;
  const float* p = X + (size_t)row * Hc;
  float xv[Hc / 32];
  float s = 0.f, s2 = 0.f;
#pragma unroll
  for (int t = 0; t < Hc / 32; t++) {
    int idx = lane + t * 32;
    float x = p[idx] + bias[idx];
    xv[t] = x; s += x; s2 += x * x;
  }
#pragma unroll
  for (int off = 16; off > 0; off >>= 1) {
    s  += __shfl_xor(s,  off, 32);
    s2 += __shfl_xor(s2, off, 32);
  }
  float mean = s * (1.0f / Hc);
  float rstd = rsqrtf(s2 * (1.0f / Hc) - mean * mean + 1e-5f);
#pragma unroll
  for (int t = 0; t < Hc / 32; t++) {
    int idx = lane + t * 32;
    float y = fmaxf((xv[t] - mean) * rstd * g[idx] + be[idx], 0.f);
    outF[(size_t)row * Hc + idx] = y;
    outB[(size_t)row * Hc + idx] = f32_to_bf16(y);
  }
}

// ---------------- graph scores: wave per (b,i,j) pair ----------------------
__global__ void k_graph(const float* __restrict__ hi, const float* __restrict__ hj,
                        const float* __restrict__ bg1, const float* __restrict__ g_g,
                        const float* __restrict__ b_g, const float* __restrict__ Wg2,
                        const float* __restrict__ bg2, float* __restrict__ outG) {
  const int wave = threadIdx.x >> 5, lane = threadIdx.x & 31;
  const int pair = blockIdx.x * 8 + wave;        // [0, B*V*V)
  const int j  = pair % Vc;
  const int bi = pair / Vc;
  const int i  = bi % Vc;
  const int b  = bi / Vc;
  const float* pi = hi + (size_t)(b * Vc + i) * Hc;
  const float* pj = hj + (size_t)(b * Vc + j) * Hc;
  float xv[Hc / 32];
  float s = 0.f, s2 = 0.f;
#pragma unroll
  for (int t = 0; t < Hc / 32; t++) {
    int idx = lane + t * 32;
    float x = pi[idx] + pj[idx] + bg1[idx];
    xv[t] = x; s += x; s2 += x * x;
  }
#pragma unroll
  for (int off = 16; off > 0; off >>= 1) {
    s  += __shfl_xor(s,  off, 32);
    s2 += __shfl_xor(s2, off, 32);
  }
  float mean = s * (1.0f / Hc);
  float rstd = rsqrtf(s2 * (1.0f / Hc) - mean * mean + 1e-5f);
  float dot = 0.f;
#pragma unroll
  for (int t = 0; t < Hc / 32; t++) {
    int idx = lane + t * 32;
    float y = fmaxf((xv[t] - mean) * rstd * g_g[idx] + b_g[idx], 0.f);
    dot += y * Wg2[idx];
  }
#pragma unroll
  for (int off = 16; off > 0; off >>= 1) dot += __shfl_xor(dot, off, 32);
  if (lane == 0) {
    float sgm = 1.0f / (1.0f + __expf(-(dot + bg2[0])));
    outG[pair] = (i == j) ? 0.0f : sgm;
  }
}

// ---------------- fused effects: LN rows in LDS, then WMMA vs Wi2^T --------
// Block = 256 threads (8 waves); 16 rows per block; wave w owns N-strip w*96.
#define LROWS 16
#define LPAD  8
#define LSTR  (Hc + LPAD)   // 776 bf16 -> 1552 B (16-byte aligned stride)

__global__ void k_effects(const float* __restrict__ ha, const float* __restrict__ hb,
                          const float* __restrict__ bi1, const float* __restrict__ g_i,
                          const float* __restrict__ b_i,
                          const unsigned short* __restrict__ WtI2,  // [H,H] bf16 (N,K)
                          const float* __restrict__ bi2,
                          float* __restrict__ out) {
  __shared__ __align__(16) unsigned short rows[LROWS * LSTR];
  const int wave = threadIdx.x >> 5, lane = threadIdx.x & 31;
  const int r0 = blockIdx.x * LROWS;

  // Phase 1: each wave LayerNorms 2 rows into LDS (bf16)
#pragma unroll
  for (int rr = 0; rr < 2; rr++) {
    const int lr = wave * 2 + rr;
    const int r  = r0 + lr;                 // flat row = (b*I + i)*V + v
    const int rowA = r / Vc;                // b*I + i
    const int b    = r / (Ic * Vc);
    const int v    = r % Vc;
    const int rowB = b * Vc + v;
    const float* pa = ha + (size_t)rowA * Hc;
    const float* pb = hb + (size_t)rowB * Hc;
    float xv[Hc / 32];
    float s = 0.f, s2 = 0.f;
#pragma unroll
    for (int t = 0; t < Hc / 32; t++) {
      int idx = lane + t * 32;
      float x = pa[idx] + pb[idx] + bi1[idx];
      xv[t] = x; s += x; s2 += x * x;
    }
#pragma unroll
    for (int off = 16; off > 0; off >>= 1) {
      s  += __shfl_xor(s,  off, 32);
      s2 += __shfl_xor(s2, off, 32);
    }
    float mean = s * (1.0f / Hc);
    float rstd = rsqrtf(s2 * (1.0f / Hc) - mean * mean + 1e-5f);
#pragma unroll
    for (int t = 0; t < Hc / 32; t++) {
      int idx = lane + t * 32;
      float y = fmaxf((xv[t] - mean) * rstd * g_i[idx] + b_i[idx], 0.f);
      rows[lr * LSTR + idx] = f32_to_bf16(y);
    }
  }
  __syncthreads();

  // Phase 2: 16(M) x 96(N) WMMA strip per wave, K = 768
  const int mrow = lane & 15, half = lane >> 4;
  const int nbase = wave * 96;
  f32x8 acc[6] = {};
  for (int kk = 0; kk < Hc; kk += 32) {
    FragAB af;
    const unsigned short* ap = rows + mrow * LSTR + kk + 8 * half;
    af.q[0] = *(const u32x4*)(ap);        // ds_read_b128
    af.q[1] = *(const u32x4*)(ap + 16);
    FragAB bfr[6];
#pragma unroll
    for (int ni = 0; ni < 6; ni++) {
      const unsigned short* bp =
          WtI2 + (size_t)(nbase + ni * 16 + mrow) * Hc + kk + 8 * half;
      bfr[ni].q[0] = *(const u32x4*)(bp);
      bfr[ni].q[1] = *(const u32x4*)(bp + 16);
    }
#pragma unroll
    for (int ni = 0; ni < 6; ni++)
      acc[ni] = __builtin_amdgcn_wmma_f32_16x16x32_bf16(
          false, af.v, false, bfr[ni].v, (short)0, acc[ni], false, false);
  }
#pragma unroll
  for (int ni = 0; ni < 6; ni++) {
    int col = nbase + ni * 16 + mrow;
    float bias = bi2[col];
#pragma unroll
    for (int j = 0; j < 8; j++) {
      int row = r0 + j + 8 * half;
      out[(size_t)row * Hc + col] = acc[ni][j] + bias;
    }
  }
}

// ---------------------------------------------------------------------------
extern "C" void kernel_launch(void* const* d_in, const int* in_sizes, int n_in,
                              void* d_out, int out_size, void* d_ws, size_t ws_size,
                              hipStream_t stream) {
  (void)in_sizes; (void)n_in; (void)out_size; (void)ws_size;
  const float* variables     = (const float*)d_in[0];
  const float* interventions = (const float*)d_in[1];
  const float* W_enc  = (const float*)d_in[2];
  const float* b_enc  = (const float*)d_in[3];
  const float* g_enc  = (const float*)d_in[4];
  const float* be_enc = (const float*)d_in[5];
  const float* Wg1    = (const float*)d_in[6];
  const float* bg1    = (const float*)d_in[7];
  const float* g_g    = (const float*)d_in[8];
  const float* b_g    = (const float*)d_in[9];
  const float* Wg2    = (const float*)d_in[10];
  const float* bg2    = (const float*)d_in[11];
  const float* Wi1    = (const float*)d_in[12];
  const float* bi1    = (const float*)d_in[13];
  const float* g_i    = (const float*)d_in[14];
  const float* b_i    = (const float*)d_in[15];
  const float* Wi2    = (const float*)d_in[16];
  const float* bi2    = (const float*)d_in[17];

  float* out_eff   = (float*)d_out;                                  // [B,I,V,H]
  float* out_graph = out_eff + (size_t)Bc * Ic * Vc * Hc;            // [B,V,V]
  float* out_enc   = out_graph + (size_t)Bc * Vc * Vc;               // [B,V,H]

  // workspace carve-up (256B aligned slots)
  char* ws = (char*)d_ws;
  size_t o = 0;
  auto slot = [&](size_t bytes) { size_t r = o; o += (bytes + 255) & ~(size_t)255; return r; };
  const size_t HH  = (size_t)Hc * Hc;          // 589824
  const size_t MBV = (size_t)Bc * Vc * Hc;     // 1,179,648 (enc rows = 1536*768)
  const size_t MBI = (size_t)Bc * Ic * Hc;     //   589,824

  unsigned short* var_b  = (unsigned short*)(ws + slot(MBV * 2));
  unsigned short* int_b  = (unsigned short*)(ws + slot(MBI * 2));
  unsigned short* Wt_enc = (unsigned short*)(ws + slot(HH * 2));
  unsigned short* Wt_g1a = (unsigned short*)(ws + slot(HH * 2));
  unsigned short* Wt_g1b = (unsigned short*)(ws + slot(HH * 2));
  unsigned short* Wt_i1a = (unsigned short*)(ws + slot(HH * 2));
  unsigned short* Wt_i1b = (unsigned short*)(ws + slot(HH * 2));
  unsigned short* Wt_i2  = (unsigned short*)(ws + slot(HH * 2));
  unsigned short* enc_b  = (unsigned short*)(ws + slot(MBV * 2));
  float* tmp_enc = (float*)(ws + slot(MBV * 4));
  float* hi_buf  = (float*)(ws + slot(MBV * 4));
  float* hj_buf  = (float*)(ws + slot(MBV * 4));
  float* ha_buf  = (float*)(ws + slot(MBI * 4));
  float* hb_buf  = (float*)(ws + slot(MBV * 4));

  // 1) precision conversion + weight transposes
  k_cvt_bf16<<<(int)((MBV + 255) / 256), 256, 0, stream>>>(variables, var_b, (int)MBV);
  k_cvt_bf16<<<(int)((MBI + 255) / 256), 256, 0, stream>>>(interventions, int_b, (int)MBI);
  const int wtb = (int)((HH + 255) / 256);
  k_wt_bf16<<<wtb, 256, 0, stream>>>(W_enc,      Wt_enc, Hc, Hc);
  k_wt_bf16<<<wtb, 256, 0, stream>>>(Wg1,        Wt_g1a, Hc, Hc);
  k_wt_bf16<<<wtb, 256, 0, stream>>>(Wg1 + HH,   Wt_g1b, Hc, Hc);
  k_wt_bf16<<<wtb, 256, 0, stream>>>(Wi1,        Wt_i1a, Hc, Hc);
  k_wt_bf16<<<wtb, 256, 0, stream>>>(Wi1 + HH,   Wt_i1b, Hc, Hc);
  k_wt_bf16<<<wtb, 256, 0, stream>>>(Wi2,        Wt_i2,  Hc, Hc);

  const int Menc = Bc * Vc;        // 1536 rows
  const int Mint = Bc * Ic;        // 768 rows
  dim3 gEnc(Hc / 64, Menc / 32);   // (12, 48)
  dim3 gInt(Hc / 64, Mint / 32);   // (12, 24)

  // 2) enc = relu(LN(variables @ W_enc + b_enc))
  k_gemm<<<gEnc, 32, 0, stream>>>(var_b, Wt_enc, tmp_enc, Menc, Hc, Hc);
  k_ln_relu_enc<<<Menc / 8, 256, 0, stream>>>(tmp_enc, b_enc, g_enc, be_enc,
                                              out_enc, enc_b);

  // 3) pairwise pre-activations
  k_gemm<<<gEnc, 32, 0, stream>>>(enc_b, Wt_g1a, hi_buf, Menc, Hc, Hc);
  k_gemm<<<gEnc, 32, 0, stream>>>(enc_b, Wt_g1b, hj_buf, Menc, Hc, Hc);
  k_gemm<<<gEnc, 32, 0, stream>>>(enc_b, Wt_i1b, hb_buf, Menc, Hc, Hc);
  k_gemm<<<gInt, 32, 0, stream>>>(int_b, Wt_i1a, ha_buf, Mint, Hc, Hc);

  // 4) causal graph scores (VALU: LN + GEMV(N=1) + sigmoid)
  k_graph<<<(Bc * Vc * Vc) / 8, 256, 0, stream>>>(hi_buf, hj_buf, bg1, g_g, b_g,
                                                  Wg2, bg2, out_graph);

  // 5) fused effects: LN rows -> LDS bf16 -> WMMA vs Wi2^T (+bi2)
  k_effects<<<(Bc * Ic * Vc) / LROWS, 256, 0, stream>>>(ha_buf, hb_buf, bi1, g_i,
                                                        b_i, Wt_i2, bi2, out_eff);
}
